// Attention3_55533927137405
// MI455X (gfx1250) — matrix operand order repaired and checked by reading, not compiled, not run on hardware
//
#include <hip/hip_runtime.h>
#include <hip/hip_bf16.h>

typedef __attribute__((ext_vector_type(16))) _Float16 v16h;
typedef __attribute__((ext_vector_type(8)))  float    v8f;
typedef __attribute__((ext_vector_type(4)))  unsigned int u32x4;
typedef __attribute__((ext_vector_type(8)))  int      i32x8;
typedef __attribute__((ext_vector_type(4)))  int      i32x4;

#if __has_builtin(__builtin_amdgcn_tensor_load_to_lds) && __has_builtin(__builtin_amdgcn_s_wait_tensorcnt)
#define HAS_TDM 1
#else
#define HAS_TDM 0
#endif

// Problem constants (B=1 fixed)
static constexpr int  C    = 32;
static constexpr int  CH   = 16;          // half channels
static constexpr int  D    = 16;
static constexpr int  H    = 128;
static constexpr int  W    = 128;
static constexpr int  HW   = H * W;       // 16384
static constexpr long N    = (long)D * HW; // 262144
static constexpr int  C5   = 160;
static constexpr int  HEADS = 8;
static constexpr long HWC  = N / HEADS;   // 32768

// ---------------------------------------------------------------------------
// Laplacian pyramid (depthwise gaussian conv + align-corners bilinear resize)
// ---------------------------------------------------------------------------
__global__ void k_init_copy(const float* __restrict__ x, float* __restrict__ tot,
                            float* __restrict__ prev) {
  long i = (long)blockIdx.x * blockDim.x + threadIdx.x;
  if (i >= (long)C * N) return;
  float v = x[i];
  tot[i]  = v;
  prev[i] = v;
}

__global__ void k_lap_conv(const float* __restrict__ prev, float* __restrict__ conv,
                           int ks, float sigma, int convH, int convW) {
  int plane = blockIdx.x;                   // 512 planes of 128x128
  const float* p = prev + (long)plane * HW;
  float* o = conv + (long)plane * HW;
  float g[10];
  float s = 0.f;
  for (int i = 0; i < ks; ++i) {
    float d = (float)i - (ks - 1) * 0.5f;
    g[i] = expf(-(d * d) / (2.f * sigma * sigma));
    s += g[i];
  }
  for (int i = 0; i < ks; ++i) g[i] /= s;
  int total = convH * convW;
  for (int idx0 = threadIdx.x; idx0 < total; idx0 += blockDim.x) {
    int oy = idx0 / convW, ox = idx0 % convW;
    float acc = 0.f;
    for (int a = 0; a < ks; ++a) {
      const float* rowp = p + (oy + a) * W + ox;
      float r = 0.f;
      for (int b = 0; b < ks; ++b) r += g[b] * rowp[b];
      acc += g[a] * r;
    }
    o[idx0] = acc;
  }
}

__global__ void k_lap_resize(const float* __restrict__ conv, float* __restrict__ prevIO,
                             float* __restrict__ tot, int convH, int convW) {
  int plane = blockIdx.x;
  const float* cv = conv + (long)plane * HW;
  float* pv = prevIO + (long)plane * HW;
  float* tv = tot + (long)plane * HW;
  float sy = (convH - 1) / (float)(H - 1);
  float sx = (convW - 1) / (float)(W - 1);
  for (int p = threadIdx.x; p < HW; p += blockDim.x) {
    int y = p / W, x = p % W;
    float cy = y * sy, cx = x * sx;
    int ly = min((int)cy, convH - 2); float fy = cy - ly;
    int lx = min((int)cx, convW - 2); float fx = cx - lx;
    float v00 = cv[ly * convW + lx],       v01 = cv[ly * convW + lx + 1];
    float v10 = cv[(ly + 1) * convW + lx], v11 = cv[(ly + 1) * convW + lx + 1];
    float g0 = v00 * (1.f - fy) + v10 * fy;
    float g1 = v01 * (1.f - fy) + v11 * fy;
    float G  = g0 * (1.f - fx) + g1 * fx;
    tv[p] += pv[p] - G;
    pv[p]  = G;
  }
}

// ---------------------------------------------------------------------------
// Directional sorts on first 16 channels (D insertion; H/W bitonic in LDS)
// ---------------------------------------------------------------------------
__global__ void k_sort_d(const float* __restrict__ lap, float* __restrict__ xs,
                         unsigned char* __restrict__ idxd) {
  long tid = (long)blockIdx.x * blockDim.x + threadIdx.x;
  if (tid >= (long)CH * HW) return;
  int c = (int)(tid / HW);
  int p = (int)(tid % HW);
  float v[16]; int id[16];
  for (int d = 0; d < D; ++d) { v[d] = lap[((long)c * D + d) * HW + p]; id[d] = d; }
  for (int i = 1; i < D; ++i) {
    float kv = v[i]; int ki = id[i]; int j = i - 1;
    while (j >= 0 && v[j] > kv) { v[j + 1] = v[j]; id[j + 1] = id[j]; --j; }
    v[j + 1] = kv; id[j + 1] = ki;
  }
  for (int d = 0; d < D; ++d) {
    long o = ((long)c * D + d) * HW + p;
    xs[o] = v[d];
    idxd[o] = (unsigned char)id[d];
  }
}

__device__ __forceinline__ unsigned int f2key(float f) {
  unsigned int u = __float_as_uint(f);
  u ^= (u >> 31) ? 0xFFFFFFFFu : 0x80000000u;
  return u;
}
__device__ __forceinline__ float key2f(unsigned int u) {
  u ^= (u >> 31) ? 0x80000000u : 0xFFFFFFFFu;
  return __uint_as_float(u);
}

// generic bitonic-128 along a strided axis
__global__ void k_sort_axis128(float* __restrict__ xs, unsigned char* __restrict__ idxo,
                               int strideSel /*0: H-axis (stride W), 1: W-axis (stride 1)*/) {
  __shared__ unsigned long long s[128];
  int blk = blockIdx.x;            // 16*16*128 columns/rows
  int t = threadIdx.x;
  long base; long stride;
  if (strideSel == 0) {            // sort along H at fixed (c,d,w)
    int w = blk % W; int rest = blk / W; int d = rest % D; int c = rest / D;
    base = ((long)c * D + d) * HW + w;
    stride = W;
  } else {                         // sort along W at fixed (c,d,h)
    int h = blk % H; int rest = blk / H; int d = rest % D; int c = rest / D;
    base = ((long)c * D + d) * HW + (long)h * W;
    stride = 1;
  }
  float val = xs[base + (long)t * stride];
  s[t] = ((unsigned long long)f2key(val) << 32) | (unsigned int)t;
  __syncthreads();
  for (int k = 2; k <= 128; k <<= 1) {
    for (int j = k >> 1; j > 0; j >>= 1) {
      int ixj = t ^ j;
      if (ixj > t) {
        bool up = (t & k) == 0;
        unsigned long long A = s[t], B = s[ixj];
        if ((A > B) == up) { s[t] = B; s[ixj] = A; }
      }
      __syncthreads();
    }
  }
  unsigned long long e = s[t];
  xs[base + (long)t * stride]   = key2f((unsigned int)(e >> 32));
  idxo[base + (long)t * stride] = (unsigned char)(e & 0xFFu);
}

// ---------------------------------------------------------------------------
// Pack [sorted_half ; raw_half] -> f16 GEMM operand; f32->f16 weight convert
// ---------------------------------------------------------------------------
__global__ void k_pack_xcat(const float* __restrict__ xs, const float* __restrict__ lap,
                            _Float16* __restrict__ xc) {
  long i = (long)blockIdx.x * blockDim.x + threadIdx.x;
  if (i >= (long)C * N) return;
  int c = (int)(i / N); long p = i % N;
  float v = (c < CH) ? xs[(long)c * N + p] : lap[(long)c * N + p];
  xc[i] = (_Float16)v;
}

__global__ void k_cvt_f16(const float* __restrict__ src, _Float16* __restrict__ dst, int n) {
  int i = blockIdx.x * blockDim.x + threadIdx.x;
  if (i < n) dst[i] = (_Float16)src[i];
}

// ---------------------------------------------------------------------------
// WMMA GEMM: Y[M x N] = Wt[M x 32] @ X[32 x N], f16 in / f32 out, K=32 -> 1 wmma.
// B tile (32 x 128 cols, 8KB) staged in LDS: via Tensor Data Mover when the
// toolchain exposes it (tensor_load_to_lds + s_wait_tensorcnt), else via
// coalesced b128 cooperative copies.
// ---------------------------------------------------------------------------
__global__ void k_gemm_k32(const _Float16* __restrict__ Wt, const _Float16* __restrict__ X,
                           float* __restrict__ Y) {
  __shared__ __align__(16) _Float16 tileB[32 * 128];
  int wave = threadIdx.x >> 5;
  int lane = threadIdx.x & 31;
  int m0 = blockIdx.y * 16;
  long col0 = (long)blockIdx.x * 128;       // this block's 128-column slab
  long n0 = col0 + wave * 16;

#if HAS_TDM
  if (threadIdx.x == 0) {
    // 2D D#: tile = 128 (dim0, contiguous) x 32 rows, elem 2B, row stride N.
    unsigned long long ga = (unsigned long long)(size_t)(X + col0);
    unsigned ldsa = (unsigned)(size_t)tileB;   // LDS byte offset of tile
    u32x4 g0 = (u32x4)0u;
    g0[0] = 1u;                                // count=1 (valid descriptor)
    g0[1] = ldsa;                              // lds_addr
    g0[2] = (unsigned)ga;                      // global_addr[31:0]
    g0[3] = (unsigned)((ga >> 32) & 0x01FFFFFFull) | 0x80000000u; // [56:32] | type=2
    i32x8 g1 = (i32x8)0;
    g1[0] = 1 << 16;                           // workgroup_mask=0, data_size=2B
    g1[1] = (int)((N & 0xFFFF) << 16);         // tensor_dim0[15:0]
    g1[2] = (int)((N >> 16) & 0xFFFF) | (32 << 16); // tensor_dim0[31:16] | tensor_dim1 lo
    g1[3] = 128 << 16;                         // tensor_dim1 hi | tile_dim0=128
    g1[4] = 32;                                // tile_dim1=32 (tile_dim2=0)
    g1[5] = (int)N;                            // tensor_dim0_stride[31:0]
    i32x4 g2 = (i32x4)0, g3 = (i32x4)0;
#if __clang_major__ >= 23
    i32x8 g4 = (i32x8)0;
    __builtin_amdgcn_tensor_load_to_lds(g0, g1, g2, g3, g4, 0);
#else
    __builtin_amdgcn_tensor_load_to_lds(g0, g1, g2, g3, 0);
#endif
  }
  __builtin_amdgcn_s_wait_tensorcnt(0);
  __syncthreads();
#else
  // cooperative coalesced copy: 512 16B segments, 2 per thread
  for (int ch = threadIdx.x; ch < 512; ch += 256) {
    int row = ch >> 4, seg = ch & 15;
    *(uint4*)(tileB + row * 128 + seg * 8) =
        *(const uint4*)(X + (long)row * N + col0 + seg * 8);
  }
  __syncthreads();
#endif

  int row = lane & 15;
  int hi  = lane >> 4;
  int kb0 = hi ? 8 : 0;
  int kb1 = kb0 + 16;
  int kc  = hi ? 16 : 0;
  v16h a, b;
#pragma unroll
  for (int t = 0; t < 8; ++t) a[t]     = Wt[(m0 + row) * 32 + kb0 + t];
#pragma unroll
  for (int t = 0; t < 8; ++t) a[8 + t] = Wt[(m0 + row) * 32 + kb1 + t];
#pragma unroll
  for (int t = 0; t < 16; ++t) b[t] = tileB[(t < 8 ? kb0 + t : kb1 + t - 8) * 128 + wave * 16 + row];
  v8f c = {};
  c = __builtin_amdgcn_wmma_f32_16x16x32_f16(false, a, false, b, (short)0, c, false, false);
#pragma unroll
  for (int r = 0; r < 8; ++r)
    Y[(long)(m0 + r + hi * 8) * N + n0 + row] = c[r];
}

// ---------------------------------------------------------------------------
// Depthwise 3x3x3 conv on 160 channels, zero pad 1
// ---------------------------------------------------------------------------
__global__ void k_dwconv(const float* __restrict__ in, const float* __restrict__ wt,
                         float* __restrict__ out) {
  long i = (long)blockIdx.x * blockDim.x + threadIdx.x;
  if (i >= (long)C5 * N) return;
  int ch = (int)(i / N); long p = i % N;
  int d = (int)(p / HW); int r = (int)(p % HW); int h = r / W; int w = r % W;
  const float* wk = wt + ch * 27;
  const float* base = in + (long)ch * N;
  __builtin_prefetch(base + p + W, 0, 0);     // global_prefetch_b8
  float acc = 0.f;
  for (int kd = 0; kd < 3; ++kd) {
    int dd = d + kd - 1; if (dd < 0 || dd >= D) continue;
    for (int kh = 0; kh < 3; ++kh) {
      int hh = h + kh - 1; if (hh < 0 || hh >= H) continue;
      for (int kw = 0; kw < 3; ++kw) {
        int ww = w + kw - 1; if (ww < 0 || ww >= W) continue;
        acc += wk[(kd * 3 + kh) * 3 + kw] * base[((long)dd * H + hh) * W + ww];
      }
    }
  }
  out[i] = acc;
}

// ---------------------------------------------------------------------------
// LSD radix argsort of v (32 independent channel segments of N)
// ---------------------------------------------------------------------------
static constexpr int RAD_TILE = 4096;
static constexpr int RAD_BLKS = (int)(N / RAD_TILE);   // 64

__global__ void k_radix_init(const float* __restrict__ v, unsigned int* __restrict__ keys,
                             unsigned int* __restrict__ vals) {
  long i = (long)blockIdx.x * blockDim.x + threadIdx.x;
  if (i >= (long)C * N) return;
  int c = (int)(i / N); long p = i % N;
  keys[i] = f2key(v[(long)c * N + p]);
  vals[i] = (unsigned int)p;
}

__global__ void k_radix_zero(unsigned int* __restrict__ hist) {
  hist[blockIdx.x * 256 + threadIdx.x] = 0u;
}

__global__ void k_radix_hist(const unsigned int* __restrict__ keys,
                             unsigned int* __restrict__ hist, int shift) {
  __shared__ unsigned int hsh[256];
  int t = threadIdx.x;
  hsh[t] = 0u;
  __syncthreads();
  int c = blockIdx.y;
  long base = (long)c * N + (long)blockIdx.x * RAD_TILE;
  for (int e = t; e < RAD_TILE; e += 256) {
    unsigned int d = (keys[base + e] >> shift) & 255u;
    atomicAdd(&hsh[d], 1u);
  }
  __syncthreads();
  if (hsh[t]) atomicAdd(&hist[c * 256 + t], hsh[t]);
}

__global__ void k_radix_scan(const unsigned int* __restrict__ hist,
                             unsigned int* __restrict__ counter) {
  __shared__ unsigned int tmp[256];
  int c = blockIdx.x, t = threadIdx.x;
  unsigned int mine = hist[c * 256 + t];
  tmp[t] = mine;
  __syncthreads();
  for (int off = 1; off < 256; off <<= 1) {
    unsigned int v0 = tmp[t];
    unsigned int add = (t >= off) ? tmp[t - off] : 0u;
    __syncthreads();
    tmp[t] = v0 + add;
    __syncthreads();
  }
  counter[c * 256 + t] = tmp[t] - mine;   // exclusive base, doubles as live counter
}

__global__ void k_radix_scatter(const unsigned int* __restrict__ keys,
                                const unsigned int* __restrict__ vals,
                                unsigned int* __restrict__ keys_o,
                                unsigned int* __restrict__ vals_o,
                                unsigned int* __restrict__ counter, int shift) {
  int c = blockIdx.y;
  long base = (long)c * N + (long)blockIdx.x * RAD_TILE;
  for (int e = threadIdx.x; e < RAD_TILE; e += 256) {
    unsigned int k = keys[base + e];
    unsigned int d = (k >> shift) & 255u;
    unsigned int r = atomicAdd(&counter[c * 256 + d], 1u);
    keys_o[(long)c * N + r] = k;
    vals_o[(long)c * N + r] = vals[base + e];
  }
}

// ---------------------------------------------------------------------------
// Row L2 norms of gathered q/k (per branch/head/row), then f16 pack of q,k,v
// ---------------------------------------------------------------------------
__global__ void k_rownorm(const float* __restrict__ qkv2, const unsigned int* __restrict__ idx,
                          float* __restrict__ norms) {
  __shared__ float red[256];
  int b = blockIdx.x;                 // [t(2)][bx(2)][h1(8)][i(32)]
  int i  = b & 31;
  int h1 = (b >> 5) & 7;
  int bx = (b >> 8) & 1;
  int tq = b >> 9;                    // 0=q, 1=k
  int c  = h1 * 4 + (i >> 3);
  int h2 = i & 7;
  int choff = tq == 0 ? (bx ? 64 : 0) : (bx ? 96 : 32);
  bool box = (bx == 0);
  const float* src = qkv2 + (long)(choff + c) * N;
  const unsigned int* ix = idx + (long)c * N;
  float s = 0.f;
  for (long m = threadIdx.x; m < HWC; m += 256) {
    long p = box ? ((long)h2 * HWC + m) : (m * 8 + h2);
    float v = src[ix[p]];
    s += v * v;
  }
  red[threadIdx.x] = s;
  __syncthreads();
  for (int off = 128; off > 0; off >>= 1) {
    if (threadIdx.x < off) red[threadIdx.x] += red[threadIdx.x + off];
    __syncthreads();
  }
  if (threadIdx.x == 0) norms[b] = fmaxf(sqrtf(red[0]), 1e-12f);
}

__global__ void k_pack_qkv(const float* __restrict__ qkv2, const unsigned int* __restrict__ idx,
                           const float* __restrict__ norms,
                           _Float16* __restrict__ qp, _Float16* __restrict__ kp,
                           _Float16* __restrict__ vp) {
  int tt = blockIdx.y;                           // 0=q,1=k,2=v
  long g = (long)blockIdx.x * blockDim.x + threadIdx.x;   // 2*8*32*HWC
  long m = g % HWC; long r = g / HWC;
  int i = (int)(r % 32); r /= 32;
  int h1 = (int)(r % 8);
  int bx = (int)(r / 8);
  int c  = h1 * 4 + (i >> 3);
  int h2 = i & 7;
  bool box = (bx == 0);
  long p = box ? ((long)h2 * HWC + m) : (m * 8 + h2);
  long src = idx[(long)c * N + p];
  int choff = (tt == 0) ? (bx ? 64 : 0) : (tt == 1) ? (bx ? 96 : 32) : 128;
  float v = qkv2[(long)(choff + c) * N + src];
  if (tt < 2) v /= norms[((long)tt * 2 + bx) * 256 + h1 * 32 + i];
  long o = (((long)bx * 8 + h1) * 32 + i) * HWC + m;
  if (tt == 0) qp[o] = (_Float16)v;
  else if (tt == 1) kp[o] = (_Float16)v;
  else vp[o] = (_Float16)v;
}

// ---------------------------------------------------------------------------
// S = Q Kᵀ  (per branch/head: 32x32 from 32x32768), WMMA K-loop (contiguous)
// ---------------------------------------------------------------------------
__global__ void k_attn_qk(const _Float16* __restrict__ Q, const _Float16* __restrict__ K,
                          float* __restrict__ S) {
  int bxh = blockIdx.x;                    // bx*8 + h1, 16 blocks
  int wave = threadIdx.x >> 5;             // 4 waves -> 2x2 tile grid
  int lane = threadIdx.x & 31;
  int i0 = (wave >> 1) * 16, j0 = (wave & 1) * 16;
  const _Float16* q = Q + (long)bxh * 32 * HWC;
  const _Float16* k = K + (long)bxh * 32 * HWC;
  int row = lane & 15, hi = lane >> 4;
  int kb0 = hi ? 8 : 0, kb1 = kb0 + 16, kc = hi ? 16 : 0;
  v8f c = {};
  for (long m0 = 0; m0 < HWC; m0 += 32) {
    v16h a, b;
    const _Float16* qr = q + (long)(i0 + row) * HWC + m0;
    __builtin_prefetch(qr + 256, 0, 0);
#pragma unroll
    for (int t = 0; t < 8; ++t) a[t]     = qr[kc + t];
#pragma unroll
    for (int t = 0; t < 8; ++t) a[8 + t] = qr[kc + 8 + t];
    const _Float16* kr = k + (long)(j0 + row) * HWC + m0 + kc;
#pragma unroll
    for (int t = 0; t < 16; ++t) b[t] = kr[t];
    c = __builtin_amdgcn_wmma_f32_16x16x32_f16(false, a, false, b, (short)0, c, false, false);
  }
#pragma unroll
  for (int r = 0; r < 8; ++r)
    S[((long)bxh * 32 + i0 + r + hi * 8) * 32 + j0 + row] = c[r];
}

__global__ void k_softmax(const float* __restrict__ S, const float* __restrict__ temp,
                          _Float16* __restrict__ A16) {
  int bxh = blockIdx.x;
  int i = threadIdx.x;                      // row, 32 threads
  float tv = temp[bxh & 7];
  float e[32], s = 0.f;
  for (int j = 0; j < 32; ++j) {
    e[j] = expf(S[((long)bxh * 32 + i) * 32 + j] * tv);
    s += e[j];
  }
  float inv = 1.f / (s + 1.f);
  for (int j = 0; j < 32; ++j)
    A16[((long)bxh * 32 + i) * 32 + j] = (_Float16)(e[j] * inv);
}

// out = A(32x32) @ V(32x32768) per branch/head; V tile staged in LDS
// (32 rows x 64 cols = 4KB, coalesced loads; block = 8 waves = 2 Mtiles x 4 Ntiles)
__global__ void k_attn_ov(const _Float16* __restrict__ A16, const _Float16* __restrict__ V,
                          float* __restrict__ O) {
  __shared__ __align__(16) _Float16 vt[32 * 64];
  int bxh = blockIdx.y;
  int wave = threadIdx.x >> 5, lane = threadIdx.x & 31;
  long col0 = (long)blockIdx.x * 64;
  int mt = wave >> 2;
  int nl = wave & 3;
  long n0 = col0 + nl * 16;
  int i0 = mt * 16;
  const _Float16* a_ = A16 + (long)bxh * 32 * 32;
  const _Float16* v_ = V + (long)bxh * 32 * HWC;

  {  // cooperative coalesced V tile load: 256 x 16B segments, 1 per thread
    int rowp = threadIdx.x >> 3, seg = threadIdx.x & 7;
    *(uint4*)(vt + rowp * 64 + seg * 8) =
        *(const uint4*)(v_ + (long)rowp * HWC + col0 + seg * 8);
  }
  __syncthreads();

  int row = lane & 15, hi = lane >> 4;
  int kb0 = hi ? 8 : 0, kb1 = kb0 + 16, kc = hi ? 16 : 0;
  v16h a, b;
#pragma unroll
  for (int tt = 0; tt < 8; ++tt) a[tt]     = a_[(i0 + row) * 32 + kb0 + tt];
#pragma unroll
  for (int tt = 0; tt < 8; ++tt) a[8 + tt] = a_[(i0 + row) * 32 + kb1 + tt];
#pragma unroll
  for (int tt = 0; tt < 16; ++tt) b[tt] = vt[(tt < 8 ? kb0 + tt : kb1 + tt - 8) * 64 + nl * 16 + row];
  v8f c = {};
  c = __builtin_amdgcn_wmma_f32_16x16x32_f16(false, a, false, b, (short)0, c, false, false);
#pragma unroll
  for (int r = 0; r < 8; ++r)
    O[((long)bxh * 32 + i0 + r + hi * 8) * HWC + n0 + row] = c[r];
}

// out1*out2 in sorted domain, scatter-unsort via idx, emit f16 for proj GEMM
__global__ void k_combine(const float* __restrict__ obuf, const unsigned int* __restrict__ idx,
                          _Float16* __restrict__ y) {
  long g = (long)blockIdx.x * blockDim.x + threadIdx.x;  // 32*N
  if (g >= (long)C * N) return;
  int c = (int)(g / N); long p = g % N;
  int h1 = c >> 2, c4 = c & 3;
  long m  = p % HWC; int h2b = (int)(p / HWC);
  long m2 = p >> 3;  int h2n = (int)(p & 7);
  float o1 = obuf[(((long)0 * 8 + h1) * 32 + c4 * 8 + h2b) * HWC + m];
  float o2 = obuf[(((long)1 * 8 + h1) * 32 + c4 * 8 + h2n) * HWC + m2];
  long dst = idx[(long)c * N + p];
  y[(long)c * N + dst] = (_Float16)(o1 * o2);
}

// undo W/H/D directional sorts for first 16 channels; passthrough for rest
__global__ void k_final(const float* __restrict__ pbuf,
                        const unsigned char* __restrict__ idd,
                        const unsigned char* __restrict__ idh,
                        const unsigned char* __restrict__ idw,
                        float* __restrict__ out) {
  long g = (long)blockIdx.x * blockDim.x + threadIdx.x;
  if (g >= (long)C * N) return;
  int c = (int)(g / N); long p = g % N;
  if (c >= CH) { out[g] = pbuf[g]; return; }
  int d = (int)(p / HW); int r = (int)(p % HW); int h = r / W; int w = r % W;
  long plane = ((long)c * D + d) * HW;
  int wp = idw[plane + (long)h * W + w];
  int hp = idh[plane + (long)h * W + wp];
  int dp = idd[plane + (long)hp * W + wp];
  out[((long)c * D + dp) * HW + (long)hp * W + wp] = pbuf[g];
}

// ---------------------------------------------------------------------------
extern "C" void kernel_launch(void* const* d_in, const int* in_sizes, int n_in,
                              void* d_out, int out_size, void* d_ws, size_t ws_size,
                              hipStream_t stream) {
  const float* x        = (const float*)d_in[0];
  const float* qkv_w    = (const float*)d_in[1];
  const float* qkv_dw_w = (const float*)d_in[2];
  const float* proj_w   = (const float*)d_in[3];
  const float* temp     = (const float*)d_in[4];
  float* out = (float*)d_out;

  char* ws = (char*)d_ws;
  size_t off = 0;
  auto alloc = [&](size_t bytes) -> void* {
    void* p = ws + off;
    off = (off + bytes + 255) & ~(size_t)255;
    return p;
  };

  float* lapTot = (float*)alloc((size_t)C * N * 4);
  float* gPrev  = (float*)alloc((size_t)C * N * 4);
  float* gConv  = (float*)alloc((size_t)C * N * 4);
  float* xsbuf  = (float*)alloc((size_t)CH * N * 4);
  unsigned char* idx_d = (unsigned char*)alloc((size_t)CH * N);
  unsigned char* idx_h = (unsigned char*)alloc((size_t)CH * N);
  unsigned char* idx_w = (unsigned char*)alloc((size_t)CH * N);
  _Float16* xcat = (_Float16*)alloc((size_t)C * N * 2);
  _Float16* wq16 = (_Float16*)alloc((size_t)C5 * 32 * 2);
  _Float16* wp16 = (_Float16*)alloc((size_t)C * 32 * 2);
  float* qkv  = (float*)alloc((size_t)C5 * N * 4);   // reused for radix buffers later
  float* qkv2 = (float*)alloc((size_t)C5 * N * 4);
  unsigned int* hist    = (unsigned int*)alloc(32 * 256 * 4);
  unsigned int* counter = (unsigned int*)alloc(32 * 256 * 4);
  float* norms = (float*)alloc(2 * 2 * 8 * 32 * 4);
  _Float16* qpack = (_Float16*)alloc((size_t)2 * 8 * 32 * HWC * 2);
  _Float16* kpack = (_Float16*)alloc((size_t)2 * 8 * 32 * HWC * 2);
  _Float16* vpack = (_Float16*)alloc((size_t)2 * 8 * 32 * HWC * 2);
  float* Sbuf  = (float*)alloc((size_t)2 * 8 * 32 * 32 * 4);
  _Float16* attn16 = (_Float16*)alloc((size_t)2 * 8 * 32 * 32 * 2);
  float* obuf  = (float*)alloc((size_t)2 * 8 * 32 * HWC * 4);
  _Float16* yprod16 = (_Float16*)alloc((size_t)C * N * 2);
  float* pbuf  = (float*)alloc((size_t)C * N * 4);

  // radix ping/pong lives in the retired qkv buffer (168MB >= 4 x 33.5MB)
  unsigned int* keysA = (unsigned int*)qkv;
  unsigned int* valsA = keysA + (size_t)C * N;
  unsigned int* keysB = valsA + (size_t)C * N;
  unsigned int* valsB = keysB + (size_t)C * N;

  const long totCN = (long)C * N;

  // 1) Laplacian pyramid
  k_init_copy<<<(totCN + 255) / 256, 256, 0, stream>>>(x, lapTot, gPrev);
  for (int lvl = 0; lvl < 3; ++lvl) {
    int ks = 3 * lvl + 4;
    float sigma = 1.6f * powf(2.0f, lvl / 3.0f);
    int convH = H - ks + 1, convW = W - ks + 1;
    k_lap_conv<<<C * D, 256, 0, stream>>>(gPrev, gConv, ks, sigma, convH, convW);
    k_lap_resize<<<C * D, 256, 0, stream>>>(gConv, gPrev, lapTot, convH, convW);
  }

  // 2) directional sorts of first half channels
  k_sort_d<<<(CH * HW + 255) / 256, 256, 0, stream>>>(lapTot, xsbuf, idx_d);
  k_sort_axis128<<<CH * D * W, 128, 0, stream>>>(xsbuf, idx_h, 0);
  k_sort_axis128<<<CH * D * H, 128, 0, stream>>>(xsbuf, idx_w, 1);

  // 3) qkv pointwise projection (WMMA) then depthwise 3x3x3
  k_pack_xcat<<<(totCN + 255) / 256, 256, 0, stream>>>(xsbuf, lapTot, xcat);
  k_cvt_f16<<<(C5 * 32 + 255) / 256, 256, 0, stream>>>(qkv_w, wq16, C5 * 32);
  k_cvt_f16<<<(C * 32 + 255) / 256, 256, 0, stream>>>(proj_w, wp16, C * 32);
  k_gemm_k32<<<dim3((int)(N / 128), C5 / 16), 256, 0, stream>>>(wq16, xcat, qkv);
  k_dwconv<<<(int)(((long)C5 * N + 255) / 256), 256, 0, stream>>>(qkv, qkv_dw_w, qkv2);

  // 4) per-channel radix argsort of v (channels 128..159 of qkv2)
  k_radix_init<<<(totCN + 255) / 256, 256, 0, stream>>>(qkv2 + (size_t)128 * N, keysA, valsA);
  unsigned int* ki = keysA; unsigned int* vi = valsA;
  unsigned int* ko = keysB; unsigned int* vo = valsB;
  for (int pass = 0; pass < 4; ++pass) {
    int shift = pass * 8;
    k_radix_zero<<<32, 256, 0, stream>>>(hist);
    k_radix_hist<<<dim3(RAD_BLKS, 32), 256, 0, stream>>>(ki, hist, shift);
    k_radix_scan<<<32, 256, 0, stream>>>(hist, counter);
    k_radix_scatter<<<dim3(RAD_BLKS, 32), 256, 0, stream>>>(ki, vi, ko, vo, counter, shift);
    unsigned int* tk = ki; ki = ko; ko = tk;
    unsigned int* tv = vi; vi = vo; vo = tv;
  }
  unsigned int* idx = vi;   // 4 passes -> back in A buffers

  // 5) gather + L2-normalize + f16 pack for WMMA attention
  k_rownorm<<<1024, 256, 0, stream>>>(qkv2, idx, norms);
  {
    long pk = (long)2 * 8 * 32 * HWC;
    k_pack_qkv<<<dim3((int)(pk / 256), 3), 256, 0, stream>>>(qkv2, idx, norms,
                                                             qpack, kpack, vpack);
  }

  // 6) channel attention: S = QK^T (WMMA), softmax-ish, O = A V (WMMA)
  k_attn_qk<<<16, 128, 0, stream>>>(qpack, kpack, Sbuf);
  k_softmax<<<16, 32, 0, stream>>>(Sbuf, temp, attn16);
  k_attn_ov<<<dim3(512, 16), 256, 0, stream>>>(attn16, vpack, obuf);

  // 7) out1*out2, unsort via idx, proj GEMM (WMMA), undo directional sorts
  k_combine<<<(totCN + 255) / 256, 256, 0, stream>>>(obuf, idx, yprod16);
  k_gemm_k32<<<dim3((int)(N / 128), C / 16), 256, 0, stream>>>(wp16, yprod16, pbuf);
  k_final<<<(totCN + 255) / 256, 256, 0, stream>>>(pbuf, idx_d, idx_h, idx_w, out);

  (void)in_sizes; (void)n_in; (void)out_size; (void)ws_size;
}
